// GraphAttentionLayerMultiHead_26070451486775
// MI455X (gfx1250) — matrix operand order repaired and checked by reading, not compiled, not run on hardware
//
#include <hip/hip_runtime.h>
#include <hip/hip_bf16.h>

typedef __attribute__((ext_vector_type(16))) __bf16 v16bf;
typedef __attribute__((ext_vector_type(8)))  __bf16 v8bf;
typedef __attribute__((ext_vector_type(8)))  float  v8f;
typedef __attribute__((ext_vector_type(4)))  float  v4f;

#define BSZ     4096
#define NNODE   64
#define IN_DIM  600
#define OUTDIM  200
#define NHEADS  8
#define KPAD    608    // 600 padded to 19 * 32
#define OPAD    208    // 200 padded to 13 * 16
#define LRELU_A 0.2f
#define SM_EPS  1e-6f

// ---------------------------------------------------------------------------
// Kernel 1: one-time conversion of w1 (200x600 f32) into zero-padded bf16
// [OPAD][KPAD] in workspace. 253 KB -> L2-resident, shared by all 4096 blocks.
// ---------------------------------------------------------------------------
__global__ void gat_convert_w1(const float* __restrict__ w1,
                               __bf16* __restrict__ wb) {
    int idx = blockIdx.x * blockDim.x + threadIdx.x;
    if (idx >= OPAD * KPAD) return;
    int r = idx / KPAD;
    int c = idx - r * KPAD;
    float v = (r < OUTDIM && c < IN_DIM) ? w1[r * IN_DIM + c] : 0.0f;
    wb[idx] = (__bf16)v;
}

// ---------------------------------------------------------------------------
// Kernel 2: one workgroup (8 wave32) per batch element.
// c = data @ w1^T + b1 via v_wmma_f32_16x16x32_bf16 (two independent
// accumulator chains per tile for ILP); c stays in LDS; attention epilogue
// (c@w2^T, leaky-relu, masked softmax over N, h = c^T @ alpha, elu) on-chip.
// ---------------------------------------------------------------------------
__global__ __launch_bounds__(256) void gat_main(
    const float* __restrict__ data,   // [BSZ][NNODE][IN_DIM]
    const float* __restrict__ mask,   // [BSZ][NNODE]
    const __bf16* __restrict__ w1b,   // [OPAD][KPAD] bf16 (padded)
    const float* __restrict__ w1_bias,// [OUTDIM]
    const float* __restrict__ w2,     // [NHEADS][OUTDIM]
    const float* __restrict__ w2_bias,// [NHEADS]
    float* __restrict__ out)          // [BSZ][OUTDIM*NHEADS]
{
    __shared__ alignas(32) __bf16 sA[NNODE * KPAD];      // 77,824 B
    __shared__ alignas(16) float  sC[NNODE * OPAD];      // 53,248 B
    __shared__ float sW2[NHEADS * OUTDIM];               //  6,400 B
    __shared__ float sB2[NHEADS];
    __shared__ float sMask[NNODE];
    __shared__ float sBmat[NNODE * NHEADS];              //  2,048 B
    __shared__ float sAlpha[NNODE * NHEADS];             //  2,048 B

    const int tid = threadIdx.x;
    const int b   = blockIdx.x;
    const float* dptr = data + (size_t)b * NNODE * IN_DIM;

    // ---- stage data tile to LDS as bf16 (vectorized, non-temporal reads) ----
    for (int i = tid; i < (NNODE * IN_DIM) / 4; i += 256) {
        int e   = i * 4;
        int row = e / IN_DIM;
        int col = e - row * IN_DIM;
        v4f v = __builtin_nontemporal_load((const v4f*)dptr + i);
        __bf16* p = &sA[row * KPAD + col];
        p[0] = (__bf16)v.x; p[1] = (__bf16)v.y;
        p[2] = (__bf16)v.z; p[3] = (__bf16)v.w;
    }
    for (int i = tid; i < NNODE * (KPAD - IN_DIM); i += 256) {
        int row = i / (KPAD - IN_DIM);
        int c   = i - row * (KPAD - IN_DIM);
        sA[row * KPAD + IN_DIM + c] = (__bf16)0.0f;
    }
    for (int i = tid; i < NHEADS * OUTDIM; i += 256) sW2[i] = w2[i];
    if (tid < NHEADS) sB2[tid]   = w2_bias[tid];
    if (tid < NNODE)  sMask[tid] = mask[(size_t)b * NNODE + tid];
    __syncthreads();

    // ---- GEMM: c[64][200] = A[64][K] * w1^T, wave-tiled 16x16x32 WMMA ----
    const int lane = tid & 31;
    const int wave = tid >> 5;
    const int g    = lane >> 4;   // half-wave select
    const int r    = lane & 15;   // row (A: M, B: N, C/D: N)

    for (int t = wave; t < 4 * 13; t += 8) {
        const int mt = t & 3;
        const int nt = t >> 2;
        const int mbase = mt * 16;
        const int nbase = nt * 16;
        const __bf16* arow = &sA[(mbase + r) * KPAD];
        const __bf16* brow = &w1b[(size_t)(nbase + r) * KPAD];

        // A 16x32 bf16: lane holds M=r, K = k+16g .. k+16g+15 (contiguous)
        auto loadA = [&](int k) -> v16bf {
            return *(const v16bf*)(arow + k + 16 * g);
        };
        // B 32x16 bf16: lane holds N=r, K = k+16g .. k+16g+15 (contiguous)
        auto loadB = [&](int k) -> v16bf {
            return *(const v16bf*)(brow + k + 16 * g);
        };

        // Two independent accumulation chains: even K-steps -> acc0,
        // odd K-steps -> acc1. 608 = 9*64 + 32 (tail into acc0).
        v8f acc0 = {};
        v8f acc1 = {};
        #pragma unroll 1
        for (int k = 0; k < 576; k += 64) {
            v16bf a0 = loadA(k);
            v16bf b0 = loadB(k);
            v16bf a1 = loadA(k + 32);
            v16bf b1 = loadB(k + 32);
            acc0 = __builtin_amdgcn_wmma_f32_16x16x32_bf16(
                false, a0, false, b0, (short)0, acc0, false, false);
            acc1 = __builtin_amdgcn_wmma_f32_16x16x32_bf16(
                false, a1, false, b1, (short)0, acc1, false, false);
        }
        {
            v16bf a0 = loadA(576);
            v16bf b0 = loadB(576);
            acc0 = __builtin_amdgcn_wmma_f32_16x16x32_bf16(
                false, a0, false, b0, (short)0, acc0, false, false);
        }

        // C/D layout: element v of acc -> row mbase + v + 8g, col nbase + r
        const int ncol = nbase + r;
        const float bias = (ncol < OUTDIM) ? w1_bias[ncol] : 0.0f;
        #pragma unroll
        for (int v = 0; v < 8; ++v)
            sC[(mbase + v + 8 * g) * OPAD + ncol] = (acc0[v] + acc1[v]) + bias;
    }
    __syncthreads();

    // ---- b = c @ w2^T + b2, leaky relu -> sBmat[64][8] ----
    for (int idx = tid; idx < NNODE * NHEADS; idx += 256) {
        int rowN = idx >> 3;
        int h    = idx & 7;
        float s = sB2[h];
        const float* crow = &sC[rowN * OPAD];
        const float* wrow = &sW2[h * OUTDIM];
        #pragma unroll 4
        for (int o = 0; o < OUTDIM; ++o) s += crow[o] * wrow[o];
        s = (s >= 0.0f) ? s : LRELU_A * s;
        sBmat[idx] = s;
    }
    __syncthreads();

    // ---- masked softmax over the 64 nodes, one thread per head ----
    if (tid < NHEADS) {
        const int h = tid;
        float mx = -3.402823e38f;
        for (int n = 0; n < NNODE; ++n) {
            float x = sBmat[n * NHEADS + h] * sMask[n];
            sBmat[n * NHEADS + h] = x;
            mx = fmaxf(mx, x);
        }
        float sum = 0.0f;
        for (int n = 0; n < NNODE; ++n) {
            float e = __expf(sBmat[n * NHEADS + h] - mx) * sMask[n];
            sAlpha[n * NHEADS + h] = e;
            sum += e;
        }
        const float inv = 1.0f / (sum + SM_EPS);
        for (int n = 0; n < NNODE; ++n) sAlpha[n * NHEADS + h] *= inv;
    }
    __syncthreads();

    // ---- h[o][head] = sum_n c[n][o] * alpha[n][head], elu, store ----
    float* orow = out + (size_t)b * (OUTDIM * NHEADS);
    for (int idx = tid; idx < OUTDIM * NHEADS; idx += 256) {
        int o = idx >> 3;
        int h = idx & 7;
        float s = 0.0f;
        #pragma unroll 4
        for (int n = 0; n < NNODE; ++n)
            s += sC[n * OPAD + o] * sAlpha[n * NHEADS + h];
        s = (s > 0.0f) ? s : (__expf(s) - 1.0f);
        __builtin_nontemporal_store(s, orow + idx);  // idx == o*HEADS + h
    }
}

// ---------------------------------------------------------------------------
extern "C" void kernel_launch(void* const* d_in, const int* in_sizes, int n_in,
                              void* d_out, int out_size, void* d_ws, size_t ws_size,
                              hipStream_t stream) {
    const float* data = (const float*)d_in[0];
    const float* mask = (const float*)d_in[1];
    const float* w1w  = (const float*)d_in[2];
    const float* w1b  = (const float*)d_in[3];
    const float* w2w  = (const float*)d_in[4];
    const float* w2b  = (const float*)d_in[5];
    float* out = (float*)d_out;
    __bf16* wsb = (__bf16*)d_ws;   // needs OPAD*KPAD*2 = 252,928 bytes

    const int total = OPAD * KPAD;
    gat_convert_w1<<<(total + 255) / 256, 256, 0, stream>>>(w1w, wsb);
    gat_main<<<BSZ, 256, 0, stream>>>(data, mask, wsb, w1b, w2w, w2b, out);
}
